// RUM_feature_17497696764609
// MI455X (gfx1250) — matrix operand order repaired and checked by reading, not compiled, not run on hardware
//
#include <hip/hip_runtime.h>
#include <hip/hip_bf16.h>

typedef _Float16 half_t;
typedef __attribute__((ext_vector_type(16))) _Float16 v16h;
typedef __attribute__((ext_vector_type(4)))  _Float16 v4h;
typedef __attribute__((ext_vector_type(8)))  float    v8f;

#define NB_ITEMS 100000
#define NB_USERS 10000
#define BB 1024
#define LL 50
#define KK 20
#define DD 64
#define ALPHA_C 0.2f

#define CVT_N4      (NB_ITEMS * DD / 4)              // 1,600,000 float4 groups
#define CVT_BLOCKS  ((CVT_N4 + 255) / 256)           // 6250
#define SCAN_BLOCKS (BB / 8)                         // 128 (8 waves per block)

__device__ __forceinline__ float fast_rcp(float x) { return __builtin_amdgcn_rcpf(x); }
__device__ __forceinline__ float fast_sig(float x) { return fast_rcp(1.0f + __expf(-x)); }
__device__ __forceinline__ float fast_tanh(float x) {
    const float t = __expf(2.0f * x);
    return (t - 1.0f) * fast_rcp(t + 1.0f);
}

// ---------------------------------------------------------------------------
// Kernel 0 (fused): blocks [0, CVT_BLOCKS) convert item_table f32 -> f16
// (12.8 MB, stays L2-resident for the GEMM); blocks [CVT_BLOCKS, +128) run
// the RUM recurrence, one wave32 per batch element (8 per block).
// Fusing lets the streaming convert overlap the latency-bound scan.
// ---------------------------------------------------------------------------
__global__ __launch_bounds__(256) void prep_kernel(
        const float* __restrict__ item_table,
        const float* __restrict__ user_table,
        const float* __restrict__ memory_init,
        const int*   __restrict__ user_id,
        const int*   __restrict__ seq,
        const int*   __restrict__ seq_length,
        half_t*      __restrict__ it_h,
        half_t*      __restrict__ pu_h) {
    if (blockIdx.x < CVT_BLOCKS) {
        // ---- f32 -> f16 item table copy ----
        const int i = blockIdx.x * 256 + threadIdx.x;
        if (i < CVT_N4) {
            const float4 v = ((const float4*)item_table)[i];
            v4h h;
            h[0] = (_Float16)v.x; h[1] = (_Float16)v.y;
            h[2] = (_Float16)v.z; h[3] = (_Float16)v.w;
            ((v4h*)it_h)[i] = h;
        }
        return;
    }

    // ---- RUM memory scan + readout: one wave32 per batch element ----
    // Lane l owns columns d0=l, d1=l+32 of the (K=20, D=64) memory state.
    // mem_emb == state after exactly seq_length[b] steps (stacked gather).
    const int b = ((int)blockIdx.x - CVT_BLOCKS) * 8 + ((int)threadIdx.x >> 5);
    const int l = threadIdx.x & 31;

    float2 m[KK];
    const float* mb = memory_init + (size_t)b * KK * DD;
#pragma unroll
    for (int k = 0; k < KK; ++k) {
        m[k].x = mb[k * DD + l];
        m[k].y = mb[k * DD + l + 32];
    }

    int T = seq_length[b];
    if (T < 0) T = 0;
    if (T > LL) T = LL;

    const int* sb = seq + (size_t)b * LL;
    float s[KK];

    for (int t = 0; t < T; ++t) {
        const int item = sb[t];
        const float* ep = item_table + (size_t)item * DD;
        const float ex = ep[l];
        const float ey = ep[l + 32];

        // s_k = sum_d m[k][d] * e[d]  (wave32 butterfly all-reduce)
#pragma unroll
        for (int k = 0; k < KK; ++k) {
            float p = m[k].x * ex + m[k].y * ey;
#pragma unroll
            for (int off = 16; off >= 1; off >>= 1)
                p += __shfl_xor(p, off, 32);
            s[k] = p;
        }
        // softmax over k (hardware exp/rcp)
        float mx = s[0];
#pragma unroll
        for (int k = 1; k < KK; ++k) mx = fmaxf(mx, s[k]);
        float sum = 0.0f;
#pragma unroll
        for (int k = 0; k < KK; ++k) { s[k] = __expf(s[k] - mx); sum += s[k]; }
        const float inv = fast_rcp(sum);

        const float erx = fast_sig(ex), ery = fast_sig(ey);
        const float adx = fast_tanh(ex), ady = fast_tanh(ey);

#pragma unroll
        for (int k = 0; k < KK; ++k) {
            const float z = s[k] * inv;
            m[k].x = m[k].x * (1.0f - z * erx) + z * adx;
            m[k].y = m[k].y * (1.0f - z * ery) + z * ady;
        }
    }

    // readout with last item (always seq[:, L-1])
    const int item = sb[LL - 1];
    const float* ep = item_table + (size_t)item * DD;
    const float ex = ep[l];
    const float ey = ep[l + 32];
#pragma unroll
    for (int k = 0; k < KK; ++k) {
        float p = m[k].x * ex + m[k].y * ey;
#pragma unroll
        for (int off = 16; off >= 1; off >>= 1)
            p += __shfl_xor(p, off, 32);
        s[k] = p;
    }
    float mx = s[0];
#pragma unroll
    for (int k = 1; k < KK; ++k) mx = fmaxf(mx, s[k]);
    float sum = 0.0f;
#pragma unroll
    for (int k = 0; k < KK; ++k) { s[k] = __expf(s[k] - mx); sum += s[k]; }
    const float inv = fast_rcp(sum);

    float px = 0.0f, py = 0.0f;
#pragma unroll
    for (int k = 0; k < KK; ++k) {
        const float z = s[k] * inv;
        px += z * m[k].x;
        py += z * m[k].y;
    }
    const float* up = user_table + (size_t)user_id[b] * DD;
    pu_h[(size_t)b * DD + l]      = (half_t)(up[l]      + ALPHA_C * px);
    pu_h[(size_t)b * DD + l + 32] = (half_t)(up[l + 32] + ALPHA_C * py);
}

// ---------------------------------------------------------------------------
// Kernel 1: scores = pu(1024x64) @ item_table^T(64x100000), f16 WMMA, f32 acc.
// Store-bound (409.6 MB out) -> nontemporal stores; f16 item table is L2 hit.
// One wave per (16-row M tile) x (128-col strip); A reused over 8 N tiles.
// NB_ITEMS is compile-time: store row stride j*400000 B and B-tile offsets
// nt*2048 B fold into the 24-bit signed IOFFSET -> one address pair per
// clause, no per-tile 64-bit pointer math. 781/782 strips take the
// guard-free fast path (uniform scalar branch, EXEC stays all-ones).
// ---------------------------------------------------------------------------
__global__ __launch_bounds__(256) void score_gemm_kernel(
        const half_t* __restrict__ it_h,
        const half_t* __restrict__ pu_h,
        float*        __restrict__ out) {
    const int lane  = threadIdx.x & 31;
    const int wave  = threadIdx.x >> 5;
    const int mtile = blockIdx.y * 8 + wave;     // 0..63
    const int r     = lane & 15;
    const int hi    = lane >> 4;                 // lane half: 0 or 1
    const int h8    = hi * 8;

    // ---- load A (pu rows mtile*16 .. +15), reused across all N tiles ----
    // A in contiguous k order (not the ISA table order): lane = half*16 + row; element e: K = 16*hi + e
    // -> one contiguous 16-half run per operand (matches B).
    const half_t* pa = pu_h + (size_t)(mtile * 16 + r) * DD;
    v16h a0, a1;
#pragma unroll
    for (int i = 0; i < 8; ++i) {
        a0[i]     = pa[hi * 16 + i];
        a0[8 + i] = pa[hi * 16 + 8 + i];
        a1[i]     = pa[32 + hi * 16 + i];
        a1[8 + i] = pa[40 + hi * 16 + i];
    }

    const int nbase = (int)blockIdx.x * 128;
    // strip-wide bases: per-tile deltas become instruction immediates
    const half_t* pb0 = it_h + (size_t)(nbase + r) * DD + hi * 16;
    float* po = out + (size_t)(mtile * 16 + hi * 8) * NB_ITEMS + (size_t)(nbase + r);

    auto tile = [&](int nt) {
        // B in contiguous k order (same as A, not the ISA table order): element e of v16h = K = 16*hi + e ->
        // one contiguous 32 B run (item_table is row-major in n).
        const v16h b0 = *(const v16h*)(pb0 + nt * 1024);        // K = 0..31
        const v16h b1 = *(const v16h*)(pb0 + nt * 1024 + 32);   // K = 32..63
        v8f c = {};
        c = __builtin_amdgcn_wmma_f32_16x16x32_f16(
                false, a0, false, b0, (short)0, c, false, false);
        c = __builtin_amdgcn_wmma_f32_16x16x32_f16(
                false, a1, false, b1, (short)0, c, false, false);
        // C layout: VGPR j -> row (mtile*16 + hi*8 + j), col nbase + nt*16 + r
        float* p = po + nt * 16;
#pragma unroll
        for (int j = 0; j < 8; ++j)
            __builtin_nontemporal_store(c[j], p + (size_t)j * NB_ITEMS);
    };

    if (nbase + 128 <= NB_ITEMS) {
        // fast path: full strip, no guards
#pragma unroll
        for (int nt = 0; nt < 8; ++nt) tile(nt);
    } else {
        // tail strip: N % 16 == 0, so per-16-tile scalar guard suffices
#pragma unroll
        for (int nt = 0; nt < 8; ++nt)
            if (nbase + nt * 16 < NB_ITEMS) tile(nt);
    }
}

// ---------------------------------------------------------------------------
extern "C" void kernel_launch(void* const* d_in, const int* in_sizes, int n_in,
                              void* d_out, int out_size, void* d_ws, size_t ws_size,
                              hipStream_t stream) {
    (void)in_sizes; (void)n_in; (void)out_size; (void)ws_size;

    const float* item_table  = (const float*)d_in[0];  // (100000, 64) f32
    const float* user_table  = (const float*)d_in[1];  // (10000, 64)  f32
    const float* memory_init = (const float*)d_in[2];  // (1024,20,64) f32
    const int*   user_id     = (const int*)d_in[3];    // (1024,)      i32
    const int*   seq         = (const int*)d_in[4];    // (1024,50)    i32
    const int*   seq_length  = (const int*)d_in[5];    // (1024,)      i32
    float*       scores      = (float*)d_out;          // (1024,100000) f32

    // workspace: [0, 12.8MB) f16 item table, then 128KB f16 pu
    half_t* it_h = (half_t*)d_ws;
    half_t* pu_h = (half_t*)((char*)d_ws + (size_t)NB_ITEMS * DD * sizeof(half_t));

    // 0) fused: f32->f16 table convert (streaming) + recurrence scan (latency)
    prep_kernel<<<CVT_BLOCKS + SCAN_BLOCKS, 256, 0, stream>>>(
        item_table, user_table, memory_init, user_id, seq, seq_length, it_h, pu_h);

    // 1) scores = pu @ item_table^T via f16 WMMA, f32 accumulate, NT stores
    dim3 grid((NB_ITEMS + 127) / 128, 8);
    score_gemm_kernel<<<grid, 256, 0, stream>>>(it_h, pu_h, scores);
}
